// LinearKernel_34428457845098
// MI455X (gfx1250) — matrix operand order repaired and checked by reading, not compiled, not run on hardware
//
#include <hip/hip_runtime.h>
#include <hip/hip_bf16.h>

typedef __attribute__((ext_vector_type(16))) __bf16 v16bf;
typedef __attribute__((ext_vector_type(8)))  __bf16 v8bf;
typedef __attribute__((ext_vector_type(8)))  float  v8f;
typedef __attribute__((ext_vector_type(4)))  unsigned int u32x4;
typedef __attribute__((ext_vector_type(8)))  int i32x8;
typedef __attribute__((ext_vector_type(4)))  int i32x4;

#define N_PTS 9216            // 96*96
#define HW 96
#define NCOL 256              // B*Cout
#define KC 32                 // K per chunk (bf16 WMMA K)
#define NCHUNK (N_PTS / KC)   // 288 (even)
#define ROWB 80               // padded LDS bytes per column: 64 data + 16 pad

// round-half-up f32->bf16 pair pack (inputs are finite)
__device__ __forceinline__ unsigned pack_bf16(float lo, float hi) {
  unsigned bl = __float_as_uint(lo), bh = __float_as_uint(hi);
  return ((bh + 0x8000u) & 0xffff0000u) | ((bl + 0x8000u) >> 16);
}

__device__ __forceinline__ float4 dist4(float4 X, float4 Y, float xr, float yr) {
  float4 r;
  float ax = X.x - xr, ay = Y.x - yr;
  float bx = X.y - xr, by = Y.y - yr;
  float cx = X.z - xr, cy = Y.z - yr;
  float dx = X.w - xr, dy = Y.w - yr;
  r.x = __builtin_amdgcn_sqrtf(ax * ax + ay * ay);
  r.y = __builtin_amdgcn_sqrtf(bx * bx + by * by);
  r.z = __builtin_amdgcn_sqrtf(cx * cx + cy * cy);
  r.w = __builtin_amdgcn_sqrtf(dx * dx + dy * dy);
  return r;
}

// ---------------------------------------------------------------------------
// Stage 1:  Tt[col][n] = bf16( bias[o] + sum_c W[o][c] * v[b][c][n] ),
//           col = b*32+o, K-major; plus coordinate tables xs[n], ys[n].
// ---------------------------------------------------------------------------
__global__ __launch_bounds__(256) void fno_proj_kernel(
    const float* __restrict__ v, const float* __restrict__ W,
    const float* __restrict__ bias, __bf16* __restrict__ Tt,
    float* __restrict__ xs, float* __restrict__ ys)
{
  __shared__ float sv[256 * 32];          // 32KB
  const int tid = threadIdx.x;
  const int nBase = blockIdx.x * 32;

  {
    const float4* vp4 = (const float4*)(v + (size_t)tid * N_PTS + nBase);
    float4* sv4 = (float4*)(sv + tid * 32);
#pragma unroll
    for (int j = 0; j < 8; ++j) sv4[j] = vp4[j];
  }
  __syncthreads();

  const int b = tid >> 5, o = tid & 31;
  float wrow[32];
#pragma unroll
  for (int c = 0; c < 32; ++c) wrow[c] = W[o * 32 + c];
  const float bo = bias[o];

  unsigned* tp = (unsigned*)(Tt + (size_t)tid * N_PTS + nBase);
  for (int j = 0; j < 16; ++j) {          // pairs of n
    float a0 = bo, a1 = bo;
    const float* svc = sv + (b * 32) * 32 + 2 * j;
#pragma unroll
    for (int c = 0; c < 32; ++c) {
      a0 += wrow[c] * svc[c * 32];
      a1 += wrow[c] * svc[c * 32 + 1];
    }
    tp[j] = pack_bf16(a0, a1);
  }

  if (tid < 32) {
    int n = nBase + tid;
    xs[n] = (float)(n % HW) * (1.0f / 95.0f);
    ys[n] = (float)(n / HW) * (1.0f / 95.0f);
  }
}

// ---------------------------------------------------------------------------
// Stage 2: U[row][col] = sum_d dist(row,d) * Tt[col][d]
//  - K loop unrolled x2: static buffer/A selection
//  - A dist tiles in float4 regs (raw v_sqrt_f32, integer bf16 pack),
//    software-pipelined one chunk ahead
//  - B double-buffered in LDS via TENSOR_LOAD_TO_LDS (TDM), 80B/col padded;
//    2-deep register ping-pong of B fragments (loads issued 2 tiles ahead)
// Block: 256 thr = 8 waves. Wave = 16 rows x 128 cols (8 C tiles). Grid: 144.
// ---------------------------------------------------------------------------
__global__ __launch_bounds__(256) void fno_integral_kernel(
    const __bf16* __restrict__ Tt, const float* __restrict__ xs,
    const float* __restrict__ ys, float* __restrict__ out)
{
  __shared__ __align__(16) unsigned char sB[2][NCOL * ROWB];  // 2 x 20KB

  const int tid   = threadIdx.x;
  const int lane  = tid & 31;
  const int wid   = tid >> 5;       // 0..7
  const int band  = wid & 3;        // row band within block
  const int chalf = wid >> 2;       // column half
  const int m     = lane & 15;      // A row / C column within a tile
  const int h     = lane >> 4;      // lane-half -> K sub-run select
  const int rBase = blockIdx.x * 64 + band * 16;
  const float xr = xs[rBase + m];
  const float yr = ys[rBase + m];

  // ---- TDM descriptor (CDNA5 ISA) ----
  const unsigned long long gbase = (unsigned long long)(uintptr_t)Tt;
  const unsigned lds0 = (unsigned)(uintptr_t)&sB[0][0];
  const unsigned lds1 = (unsigned)(uintptr_t)&sB[1][0];

  i32x8 g1;
  g1[0] = (1 << 16)                 // data_size = 2 bytes
        | (1 << 20)                 // pad_enable
        | (3 << 22)                 // pad_interval: 16 DWORDs (64B of data)
        | (3 << 25);                // pad_amount: 4 DWORDs (16B)
  g1[1] = (N_PTS & 0xffff) << 16;                       // tensor_dim0[15:0]
  g1[2] = ((N_PTS >> 16) & 0xffff) | ((NCOL & 0xffff) << 16); // td0 hi | td1 lo
  g1[3] = (KC << 16);                                   // td1 hi=0 | tile_dim0=32
  g1[4] = NCOL;                                         // tile_dim1=256, tile_dim2=0
  g1[5] = N_PTS;                                        // tensor_dim0_stride lo32
  g1[6] = 0;
  g1[7] = 0;
  const i32x4 gz4 = {0, 0, 0, 0};
  const i32x8 gz8 = {0, 0, 0, 0, 0, 0, 0, 0};

  auto issue = [&](int chunk, unsigned ldsaddr) {
    unsigned long long ga = gbase + (unsigned long long)chunk * (KC * 2);
    u32x4 g0;
    g0[0] = 1u;                                          // count=1
    g0[1] = ldsaddr;                                     // lds_addr
    g0[2] = (unsigned)(ga & 0xffffffffu);                // global_addr[31:0]
    g0[3] = (unsigned)((ga >> 32) & 0x1ffffffu)          // global_addr[56:32]
          | (2u << 30);                                  // type = 2 ("image")
    __builtin_amdgcn_tensor_load_to_lds(g0, g1, gz4, gz4, gz8, 0);
  };

  // dist A-fragment for one K-chunk: k = {8h..8h+7} U {16+8h..23+8h}
  auto makeA = [&](int d0) -> v16bf {
    const float* xp = xs + d0 + 8 * h;
    const float* yp = ys + d0 + 8 * h;
    float4 d0a = dist4(*(const float4*)(xp),      *(const float4*)(yp),      xr, yr);
    float4 d0b = dist4(*(const float4*)(xp + 4),  *(const float4*)(yp + 4),  xr, yr);
    float4 d1a = dist4(*(const float4*)(xp + 16), *(const float4*)(yp + 16), xr, yr);
    float4 d1b = dist4(*(const float4*)(xp + 20), *(const float4*)(yp + 20), xr, yr);
    union { v16bf v; unsigned u[8]; } A;
    A.u[0] = pack_bf16(d0a.x, d0a.y);
    A.u[1] = pack_bf16(d0a.z, d0a.w);
    A.u[2] = pack_bf16(d0b.x, d0b.y);
    A.u[3] = pack_bf16(d0b.z, d0b.w);
    A.u[4] = pack_bf16(d1a.x, d1a.y);
    A.u[5] = pack_bf16(d1a.z, d1a.w);
    A.u[6] = pack_bf16(d1b.x, d1b.y);
    A.u[7] = pack_bf16(d1b.z, d1b.w);
    return A.v;
  };

  auto ldB = [&](const unsigned char* bufp, int t) -> v16bf {
    const int col = (chalf * 8 + t) * 16 + m;
    const v8bf* bp = (const v8bf*)(bufp + col * ROWB + 16 * h);
    union { v16bf v; v8bf hv[2]; } B_;
    B_.hv[0] = bp[0];                      // ds_load_b128
    B_.hv[1] = bp[2];                      // ds_load_b128
    return B_.v;
  };

#define WMMA_BF16(A_, B_, C_) \
  __builtin_amdgcn_wmma_f32_16x16x32_bf16(false, (A_), false, (B_), (short)0, (C_), false, false)

// 2-deep ping-pong: B loads are issued 2 tiles ahead of their WMMA, so each
// v_wmma waits with 2 ds_loads still in flight (s_wait_dscnt <= 2, not 0).
#define PHASE(BUFP, ACUR, PRELOAD_NEXT)                                     \
  {                                                                         \
    const unsigned char* bufp_ = (BUFP);                                    \
    v16bf p0_ = ldB(bufp_, 0);                                              \
    v16bf p1_ = ldB(bufp_, 1);                                              \
    PRELOAD_NEXT;                                                           \
    acc[0] = WMMA_BF16(ACUR, p0_, acc[0]); p0_ = ldB(bufp_, 2);             \
    acc[1] = WMMA_BF16(ACUR, p1_, acc[1]); p1_ = ldB(bufp_, 3);             \
    acc[2] = WMMA_BF16(ACUR, p0_, acc[2]); p0_ = ldB(bufp_, 4);             \
    acc[3] = WMMA_BF16(ACUR, p1_, acc[3]); p1_ = ldB(bufp_, 5);             \
    acc[4] = WMMA_BF16(ACUR, p0_, acc[4]); p0_ = ldB(bufp_, 6);             \
    acc[5] = WMMA_BF16(ACUR, p1_, acc[5]); p1_ = ldB(bufp_, 7);             \
    acc[6] = WMMA_BF16(ACUR, p0_, acc[6]);                                  \
    acc[7] = WMMA_BF16(ACUR, p1_, acc[7]);                                  \
  }

  if (wid == 0) {                    // prologue: fetch chunk 0
    issue(0, lds0);
    __builtin_amdgcn_s_wait_tensorcnt(0);
  }
  __syncthreads();

  v8f acc[8] = {};
  v16bf A0 = makeA(0), A1;

  for (int c = 0; c < NCHUNK; c += 2) {
    // -------- even phase: consume sB[0] / A0, fetch chunk c+1 -> sB[1] ----
    if (wid == 0) issue(c + 1, lds1);              // c+1 <= 287 always
    PHASE(sB[0], A0, (A1 = makeA((c + 1) * KC)));
    if (wid == 0) __builtin_amdgcn_s_wait_tensorcnt(0);
    __syncthreads();

    // -------- odd phase: consume sB[1] / A1, fetch chunk c+2 -> sB[0] -----
    const bool more = (c + 2) < NCHUNK;
    if (wid == 0 && more) issue(c + 2, lds0);
    PHASE(sB[1], A1, if (more) A0 = makeA((c + 2) * KC));
    if (wid == 0) __builtin_amdgcn_s_wait_tensorcnt(0);
    __syncthreads();
  }

  // ---- C layout: lane col = m, VGPR j holds row rBase + 8h + j ----
#pragma unroll
  for (int t = 0; t < 8; ++t) {
    const int col = (chalf * 8 + t) * 16 + m;
    float* op = out + (size_t)col * N_PTS + rBase + 8 * h;
    float4 w0 = make_float4(acc[t][0], acc[t][1], acc[t][2], acc[t][3]);
    float4 w1 = make_float4(acc[t][4], acc[t][5], acc[t][6], acc[t][7]);
    *(float4*)op = w0;
    *(float4*)(op + 4) = w1;
  }
}

extern "C" void kernel_launch(void* const* d_in, const int* in_sizes, int n_in,
                              void* d_out, int out_size, void* d_ws, size_t ws_size,
                              hipStream_t stream) {
  const float* v    = (const float*)d_in[0];  // (8,32,96,96)
  const float* W    = (const float*)d_in[1];  // (32,32)
  const float* bias = (const float*)d_in[2];  // (32,)
  float* out = (float*)d_out;                 // (8,32,96,96)

  __bf16* Tt = (__bf16*)d_ws;                                   // 256*9216 bf16
  float*  xs = (float*)((char*)d_ws + (size_t)NCOL * N_PTS * 2);
  float*  ys = xs + N_PTS;

  fno_proj_kernel<<<N_PTS / 32, 256, 0, stream>>>(v, W, bias, Tt, xs, ys);
  fno_integral_kernel<<<N_PTS / 64, 256, 0, stream>>>(Tt, xs, ys, out);
}
